// MoEMLP_11390253269162
// MI455X (gfx1250) — matrix operand order repaired and checked by reading, not compiled, not run on hardware
//
#include <hip/hip_runtime.h>

// ---------------------------------------------------------------------------
// MoE MLP (top-2 of 8 experts), MI455X / gfx1250, bf16 WMMA with f32 accum.
// Shapes fixed per reference: B=4,T=1024,D=1024,E=8,H=2048,K=2.
// M=32 tokens per block (halves L2 weight traffic vs M=16), 16 waves/block.
// ---------------------------------------------------------------------------

typedef __attribute__((ext_vector_type(16))) __bf16 v16bf;
typedef __attribute__((ext_vector_type(8)))  __bf16 bf16x8;
typedef __attribute__((ext_vector_type(4)))  __bf16 bf16x4;
typedef __attribute__((ext_vector_type(8)))  float  v8f;
typedef int v4i_vs __attribute__((vector_size(16)));   // matches async-LDS builtin param

constexpr int D_ = 1024;
constexpr int E_ = 8;
constexpr int H_ = 2048;
constexpr int N_ = 4 * 1024;        // B*T tokens
constexpr int SLOTS = N_ * 2;       // top-2
constexpr int TILE_M = 32;
constexpr int TILES_MAX = SLOTS / TILE_M; // 256 (worst case one expert gets all)

#if defined(__HIP_DEVICE_COMPILE__) && \
    __has_builtin(__builtin_amdgcn_global_load_async_to_lds_b128) && \
    __has_builtin(__builtin_amdgcn_s_wait_asynccnt)
#define USE_ASYNC_LDS 1
#define AS1 __attribute__((address_space(1)))
#define AS3 __attribute__((address_space(3)))
#else
#define USE_ASYNC_LDS 0
#endif

// ---------------------------------------------------------------------------
// Pre-pass: f32 -> bf16 conversions / transposes into workspace
// ---------------------------------------------------------------------------
__global__ void convert_x_kernel(const float* __restrict__ x, __bf16* __restrict__ xb) {
    size_t i = ((size_t)blockIdx.x * 256 + threadIdx.x) * 4;
    const float4 v = *(const float4*)(x + i);
    bf16x4 o;
    o.x = (__bf16)v.x; o.y = (__bf16)v.y; o.z = (__bf16)v.z; o.w = (__bf16)v.w;
    *(bf16x4*)(xb + i) = o;
}

// src[e][r][c] (f32) -> dst[e][c][r] (bf16); R,C multiples of 32. block=(32,8)
__global__ void transpose_bf16_kernel(const float* __restrict__ src,
                                      __bf16* __restrict__ dst, int R, int C) {
    __shared__ float tile[32][33];
    int e = blockIdx.z;
    int r0 = blockIdx.y * 32, c0 = blockIdx.x * 32;
    const float* s = src + ((size_t)e * R + r0) * C + c0;
#pragma unroll
    for (int i = 0; i < 4; ++i)
        tile[threadIdx.y + i * 8][threadIdx.x] =
            s[(size_t)(threadIdx.y + i * 8) * C + threadIdx.x];
    __syncthreads();
    __bf16* d = dst + ((size_t)e * C + c0) * R + r0;
#pragma unroll
    for (int i = 0; i < 4; ++i)
        d[(size_t)(threadIdx.y + i * 8) * R + threadIdx.x] =
            (__bf16)tile[threadIdx.x][threadIdx.y + i * 8];
}

// ---------------------------------------------------------------------------
// Router: one wave per token; logits->softmax->top2->renorm; expert histogram
// ---------------------------------------------------------------------------
__global__ void router_kernel(const float* __restrict__ x, const float* __restrict__ Wg,
                              int* __restrict__ top_i, float* __restrict__ top_w,
                              int* __restrict__ counts) {
    int lane = threadIdx.x & 31;
    int wv   = threadIdx.x >> 5;
    int t    = blockIdx.x * 8 + wv;     // 8 waves -> 8 tokens per block
    float acc[E_];
#pragma unroll
    for (int j = 0; j < E_; ++j) acc[j] = 0.f;
    const float* xr = x + (size_t)t * D_;
    for (int d = lane; d < D_; d += 32) {
        float xv = xr[d];
        const float* wr = Wg + (size_t)d * E_;
#pragma unroll
        for (int j = 0; j < E_; ++j) acc[j] += xv * wr[j];
    }
#pragma unroll
    for (int j = 0; j < E_; ++j)
#pragma unroll
        for (int off = 16; off > 0; off >>= 1)
            acc[j] += __shfl_down(acc[j], off, 32);
    if (lane == 0) {
        float m = acc[0];
#pragma unroll
        for (int j = 1; j < E_; ++j) m = fmaxf(m, acc[j]);
        float p[E_], s = 0.f;
#pragma unroll
        for (int j = 0; j < E_; ++j) { p[j] = __expf(acc[j] - m); s += p[j]; }
        float inv_s = 1.f / s;
#pragma unroll
        for (int j = 0; j < E_; ++j) p[j] *= inv_s;
        int i0 = 0;
        for (int j = 1; j < E_; ++j) if (p[j] > p[i0]) i0 = j;     // ties: lowest idx
        int i1 = (i0 == 0) ? 1 : 0;
        for (int j = 0; j < E_; ++j) if (j != i0 && p[j] > p[i1]) i1 = j;
        float w0 = p[i0], w1 = p[i1];
        float inv = 1.f / (w0 + w1 + 1e-8f);
        top_i[t * 2 + 0] = i0; top_i[t * 2 + 1] = i1;
        top_w[t * 2 + 0] = w0 * inv; top_w[t * 2 + 1] = w1 * inv;
        atomicAdd(&counts[i0], 1);
        atomicAdd(&counts[i1], 1);
    }
}

__global__ void scan_kernel(const int* __restrict__ counts,
                            int* __restrict__ bases, int* __restrict__ cursor) {
    if (threadIdx.x == 0 && blockIdx.x == 0) {
        int run = 0;
        for (int e = 0; e < E_; ++e) { bases[e] = run; cursor[e] = run; run += counts[e]; }
    }
}

__global__ void scatter_kernel(const int* __restrict__ top_i,
                               int* __restrict__ cursor, int* __restrict__ bucket) {
    int t = blockIdx.x * blockDim.x + threadIdx.x;
    if (t >= N_) return;
#pragma unroll
    for (int k = 0; k < 2; ++k) {
        int e = top_i[t * 2 + k];
        int pos = atomicAdd(&cursor[e], 1);
        bucket[pos] = (t << 1) | k;
    }
}

// ---------------------------------------------------------------------------
// Fused expert MLP: 32 tokens/block, 16 waves, bf16 WMMA, relu^2 via LDS relayout
// ---------------------------------------------------------------------------
union Frag { v16bf v; struct { bf16x8 lo, hi; } p; };

__global__ __launch_bounds__(512, 1) void moe_gemm_kernel(
    const __bf16* __restrict__ xb, const __bf16* __restrict__ w1b,
    const __bf16* __restrict__ w2b, const int* __restrict__ bucket,
    const float* __restrict__ top_w, const int* __restrict__ counts,
    const int* __restrict__ bases, float* __restrict__ out) {
    int e    = blockIdx.y;
    int tile = blockIdx.x;
    int cnt  = counts[e];
    if (tile * TILE_M >= cnt) return;
    int base = bases[e];

    __shared__ __bf16 XA[TILE_M][D_];      // 64 KB: gathered x tile (bf16)
    __shared__ __bf16 hA[TILE_M][256];     // 16 KB: activated h chunk
    __shared__ int    s_tok[TILE_M];
    __shared__ float  s_w[TILE_M];

    int tid  = threadIdx.x;
    int lane = tid & 31, wv = tid >> 5;    // 16 waves

    if (tid < TILE_M) {
        int s = tile * TILE_M + tid;
        int t; float w;
        if (s < cnt) {
            int code = bucket[base + s];
            t = code >> 1;
            w = top_w[code];
        } else {                                  // pad: duplicate row, weight 0
            int code0 = bucket[base + tile * TILE_M];
            t = code0 >> 1; w = 0.f;
        }
        s_tok[tid] = t; s_w[tid] = w;
    }
    __syncthreads();

    // Stage gathered X rows into LDS (32 rows x 1024 bf16, b128 granules)
    {
        for (int idx = tid; idx < TILE_M * (D_ / 8); idx += 512) {
            int m = idx >> 7;          // row
            int c = idx & 127;         // 8-half chunk within row
#if USE_ASYNC_LDS
            __builtin_amdgcn_global_load_async_to_lds_b128(
                (AS1 v4i_vs*)((const bf16x8*)(xb + (size_t)s_tok[m] * D_) + c),
                (AS3 v4i_vs*)((bf16x8*)&XA[m][0] + c), 0, 0);
#else
            const bf16x8* src = (const bf16x8*)(xb + (size_t)s_tok[m] * D_) + c;
            *((bf16x8*)&XA[m][0] + c) = *src;
#endif
        }
#if USE_ASYNC_LDS
        __builtin_amdgcn_s_wait_asynccnt(0);
#endif
    }
    __syncthreads();

    const int mrow  = lane & 15;
    const int koffA = (lane < 16) ? 0 : 8;   // A: K {0-7,16-23} / {8-15,24-31}
    const int koffB = (lane < 16) ? 0 : 8;   // B: K {0-7,16-23} / {8-15,24-31}

    const v8f vzero = {0.f, 0.f, 0.f, 0.f, 0.f, 0.f, 0.f, 0.f};
    v8f acc0[4], acc1[4];                     // y tiles: 4 D-tiles x 2 M-subtiles
#pragma unroll
    for (int i = 0; i < 4; ++i) { acc0[i] = vzero; acc1[i] = vzero; }

    const __bf16* w1e = w1b + (size_t)e * H_ * D_;   // [h][d] bf16
    const __bf16* w2e = w2b + (size_t)e * D_ * H_;   // [d][h] bf16

    for (int hc = 0; hc < H_ / 256; ++hc) {
        // ---- phase 1: this wave computes h[0:32, hc*256+wv*16 .. +16), K=1024.
        // One B fragment feeds both M-subtiles (2x weight reuse).
        v8f c0 = vzero, c1 = vzero;
        int n0 = hc * 256 + wv * 16;
        const __bf16* bcol = w1e + (size_t)(n0 + (lane & 15)) * D_;
        for (int k0 = 0; k0 < D_; k0 += 32) {
            Frag a0, a1, b;
            a0.p.lo = *(const bf16x8*)&XA[mrow][k0 + koffA];
            a0.p.hi = *(const bf16x8*)&XA[mrow][k0 + 16 + koffA];
            a1.p.lo = *(const bf16x8*)&XA[16 + mrow][k0 + koffA];
            a1.p.hi = *(const bf16x8*)&XA[16 + mrow][k0 + 16 + koffA];
            const __bf16* bp = bcol + k0 + koffB;
            b.p.lo = *(const bf16x8*)bp;  b.p.hi = *(const bf16x8*)(bp + 16);
            c0 = __builtin_amdgcn_wmma_f32_16x16x32_bf16(false, a0.v, false, b.v,
                                                         (short)0, c0, false, false);
            c1 = __builtin_amdgcn_wmma_f32_16x16x32_bf16(false, a1.v, false, b.v,
                                                         (short)0, c1, false, false);
        }
        // relu^2 + relayout C->A via LDS
#pragma unroll
        for (int r = 0; r < 8; ++r) {
            int m = r + ((lane < 16) ? 0 : 8);
            float v0 = fmaxf(c0[r], 0.f); v0 = v0 * v0;
            float v1 = fmaxf(c1[r], 0.f); v1 = v1 * v1;
            hA[m][wv * 16 + (lane & 15)]      = (__bf16)v0;
            hA[16 + m][wv * 16 + (lane & 15)] = (__bf16)v1;
        }
        __syncthreads();
        // ---- phase 2: y[0:32, wave's 4 D-tiles] += h_chunk @ W2 chunk (K=256).
        // One B fragment feeds both M-subtiles.
        for (int k0 = 0; k0 < 256; k0 += 32) {
            Frag a0, a1;
            a0.p.lo = *(const bf16x8*)&hA[mrow][k0 + koffA];
            a0.p.hi = *(const bf16x8*)&hA[mrow][k0 + 16 + koffA];
            a1.p.lo = *(const bf16x8*)&hA[16 + mrow][k0 + koffA];
            a1.p.hi = *(const bf16x8*)&hA[16 + mrow][k0 + 16 + koffA];
            int kg = hc * 256 + k0 + koffB;
#pragma unroll
            for (int nt = 0; nt < 4; ++nt) {
                int d = (wv * 4 + nt) * 16 + (lane & 15);
                Frag b;
                const __bf16* bp = w2e + (size_t)d * H_ + kg;
                b.p.lo = *(const bf16x8*)bp;  b.p.hi = *(const bf16x8*)(bp + 16);
                acc0[nt] = __builtin_amdgcn_wmma_f32_16x16x32_bf16(
                    false, a0.v, false, b.v, (short)0, acc0[nt], false, false);
                acc1[nt] = __builtin_amdgcn_wmma_f32_16x16x32_bf16(
                    false, a1.v, false, b.v, (short)0, acc1[nt], false, false);
            }
        }
        __syncthreads();   // hA rewritten next chunk
    }

    // Epilogue: out[t, d] += w_t * y  (exactly 2 commutative adds per element)
#pragma unroll
    for (int nt = 0; nt < 4; ++nt) {
        int d = (wv * 4 + nt) * 16 + (lane & 15);
#pragma unroll
        for (int r = 0; r < 8; ++r) {
            int m0 = r + ((lane < 16) ? 0 : 8);
            int m1 = 16 + m0;
            atomicAdd(&out[(size_t)s_tok[m0] * D_ + d], s_w[m0] * acc0[nt][r]);
            atomicAdd(&out[(size_t)s_tok[m1] * D_ + d], s_w[m1] * acc1[nt][r]);
        }
    }
}

// ---------------------------------------------------------------------------
extern "C" void kernel_launch(void* const* d_in, const int* in_sizes, int n_in,
                              void* d_out, int out_size, void* d_ws, size_t ws_size,
                              hipStream_t stream) {
    (void)in_sizes; (void)n_in; (void)ws_size;
    const float* x  = (const float*)d_in[0];   // (4,1024,1024)
    const float* Wg = (const float*)d_in[1];   // (1024,8)
    const float* W1 = (const float*)d_in[2];   // (8,1024,2048)
    const float* W2 = (const float*)d_in[3];   // (8,2048,1024)
    float* out = (float*)d_out;

    char* ws = (char*)d_ws;
    size_t off = 0;
    auto alloc = [&](size_t bytes) -> char* {
        char* p = ws + off;
        off = (off + bytes + 255) & ~(size_t)255;
        return p;
    };
    __bf16* xb     = (__bf16*)alloc((size_t)N_ * D_ * 2);        //  8 MB
    __bf16* w1b    = (__bf16*)alloc((size_t)E_ * D_ * H_ * 2);   // 32 MB  [e][h][d]
    __bf16* w2b    = (__bf16*)alloc((size_t)E_ * D_ * H_ * 2);   // 32 MB  [e][d][h]
    int*    top_i  = (int*)  alloc(SLOTS * 4);
    float*  top_w  = (float*)alloc(SLOTS * 4);
    int*    bucket = (int*)  alloc(SLOTS * 4);
    int*    counts = (int*)  alloc(E_ * 4);
    int*    bases  = (int*)  alloc(E_ * 4);
    int*    cursor = (int*)  alloc(E_ * 4);

    (void)hipMemsetAsync(out, 0, (size_t)out_size * sizeof(float), stream);
    (void)hipMemsetAsync(counts, 0, E_ * sizeof(int), stream);

    convert_x_kernel<<<(N_ * D_ / 4) / 256, 256, 0, stream>>>(x, xb);
    dim3 tb(32, 8);
    transpose_bf16_kernel<<<dim3(H_ / 32, D_ / 32, E_), tb, 0, stream>>>(W1, w1b, D_, H_);
    transpose_bf16_kernel<<<dim3(D_ / 32, H_ / 32, E_), tb, 0, stream>>>(W2, w2b, H_, D_);
    router_kernel<<<N_ / 8, 256, 0, stream>>>(x, Wg, top_i, top_w, counts);
    scan_kernel<<<1, 32, 0, stream>>>(counts, bases, cursor);
    scatter_kernel<<<(N_ + 255) / 256, 256, 0, stream>>>(top_i, cursor, bucket);
    moe_gemm_kernel<<<dim3(TILES_MAX, E_), 512, 0, stream>>>(
        xb, w1b, w2b, bucket, top_w, counts, bases, out);
}
